// dictloss_163208757659
// MI455X (gfx1250) — matrix operand order repaired and checked by reading, not compiled, not run on hardware
//
#include <hip/hip_runtime.h>

typedef __bf16 bf16_t;
typedef bf16_t v16bf __attribute__((ext_vector_type(16)));
typedef float  v8f   __attribute__((ext_vector_type(8)));

#define N_SIG   262144   // signal length N
#define KCOLS   262081   // npatches K = N - P + 1
#define NATOMS  256      // inner GEMM dimension
#define PSZ     64       // patch size P (GEMM M)
#define MROWS   512      // measurement rows
#define COLS_PER_BLOCK 128

// ---------------------------------------------------------------------------
// Kernel 1: ss_b = D[64x256] @ X[256xK] + meanY, fused with the stride-1
// sliding-window scatter:  ss_p_sum[n] += sum_p ss_b[p, n-p]  (anti-diagonal).
// 8 waves/block; wave w owns columns [k0+16w, k0+16w+16). Each wave:
//   - loads one B tile (32x16 bf16) per K-step  -> X read exactly once
//   - runs 4 stacked v_wmma_f32_16x16x32_bf16 covering all 64 rows of D
//   - scatters C (+meanY) onto an LDS diagonal accumulator via ds_add_f32
// Block then flushes 192 partial sums to global with atomicAdd (block overlap).
// ---------------------------------------------------------------------------
__global__ __launch_bounds__(256) void k_gemm_diag(
    const float* __restrict__ D, const float* __restrict__ X,
    const float* __restrict__ meanY_p, float* __restrict__ ss_p_sum)
{
    __shared__ float diag[COLS_PER_BLOCK + PSZ];   // 192 floats
    const int tid = threadIdx.x;
    for (int i = tid; i < COLS_PER_BLOCK + PSZ; i += 256) diag[i] = 0.0f;
    __syncthreads();

    const int  wave = tid >> 5;
    const int  lane = tid & 31;
    const int  lo16 = lane & 15;
    const bool hi   = lane >= 16;
    const int  k0   = blockIdx.x * COLS_PER_BLOCK;
    const int  col  = k0 + wave * 16 + lo16;       // this lane's B column (N)
    const bool colValid = col < KCOLS;
    const float meanY = meanY_p[0];

    const int bK = hi ? 16 : 0;  // B: lanes 0-15 hold K=0..15, 16-31 hold K=16..31
    const int aK = hi ? 16 : 0;  // A: lanes 0-15 hold K=0..15, 16-31 hold K=16..31

    v8f c[4] = {};               // 4 stacked 16x16 f32 accumulators (M=0..63)

    for (int kk = 0; kk < NATOMS; kk += 32) {
        // ---- B tile: X[kk+bK+j, col], zero-fill out-of-range columns ----
        v16bf b;
        #pragma unroll
        for (int j = 0; j < 16; ++j) {
            float xv = colValid ? X[(size_t)(kk + bK + j) * KCOLS + col] : 0.0f;
            b[j] = (bf16_t)xv;
        }
        // ---- 4 A tiles (rows m*16..m*16+15) and WMMAs ----
        #pragma unroll
        for (int m = 0; m < 4; ++m) {
            v16bf a;
            const int row = m * 16 + lo16;
            #pragma unroll
            for (int j = 0; j < 16; ++j) {
                // A element j -> K = aK + j
                const int ka = kk + aK + j;
                a[j] = (bf16_t)D[row * NATOMS + ka];
            }
            c[m] = __builtin_amdgcn_wmma_f32_16x16x32_bf16(
                       false, a, false, b, (short)0, c[m], false, false);
        }
    }

    // ---- anti-diagonal scatter into LDS: n_local = col_local + p ----
    if (colValid) {
        const int nloc = wave * 16 + lo16;         // column offset within block
        const int mofs = hi ? 8 : 0;               // C/D layout: hi lanes = M+8
        #pragma unroll
        for (int m = 0; m < 4; ++m) {
            #pragma unroll
            for (int r = 0; r < 8; ++r) {
                const int p = m * 16 + mofs + r;   // row index (patch offset)
                atomicAdd(&diag[nloc + p], c[m][r] + meanY);
            }
        }
    }
    __syncthreads();

    // ---- flush block-local diagonal sums to global ----
    for (int i = tid; i < COLS_PER_BLOCK + PSZ; i += 256) {
        const int n = k0 + i;
        if (n < N_SIG) {
            const float s = diag[i];
            if (s != 0.0f) atomicAdd(&ss_p_sum[n], s);
        }
    }
}

// ---------------------------------------------------------------------------
// Kernel 2: v[n] = ((lam2*ds + ss_p_sum)/(lam2+npp)) * vb + sRef
// ---------------------------------------------------------------------------
__global__ __launch_bounds__(256) void k_finalize(
    const float* __restrict__ ss_p_sum, const float* __restrict__ ds,
    const float* __restrict__ npp, const float* __restrict__ vb,
    const float* __restrict__ sRef, const float* __restrict__ lam2_p,
    float* __restrict__ v)
{
    const float lam2 = lam2_p[0];
    const int i = blockIdx.x * blockDim.x + threadIdx.x;
    if (i < N_SIG) {
        const float ssf = (lam2 * ds[i] + ss_p_sum[i]) / (lam2 + npp[i]);
        v[i] = ssf * vb[i] + sRef[i];
    }
}

// ---------------------------------------------------------------------------
// Kernel 3: Tref[m] += A[m, chunk] . v[chunk]   (bandwidth-bound GEMV)
// grid = (8 chunks of 32768 cols, 512 rows); float4 streaming loads.
// ---------------------------------------------------------------------------
__global__ __launch_bounds__(256) void k_matvec(
    const float* __restrict__ A, const float* __restrict__ v,
    float* __restrict__ Tref)
{
    __shared__ float red[256];
    const int m     = blockIdx.y;
    const int chunk = blockIdx.x;
    const float4* Ap = (const float4*)(A + (size_t)m * N_SIG + (size_t)chunk * 32768);
    const float4* vp = (const float4*)(v + (size_t)chunk * 32768);

    float sum = 0.0f;
    #pragma unroll 4
    for (int t = 0; t < 32; ++t) {
        const int idx = t * 256 + threadIdx.x;
        const float4 av = Ap[idx];
        const float4 vv = vp[idx];
        sum += av.x * vv.x + av.y * vv.y + av.z * vv.z + av.w * vv.w;
    }
    red[threadIdx.x] = sum;
    __syncthreads();
    for (int s = 128; s > 0; s >>= 1) {
        if (threadIdx.x < s) red[threadIdx.x] += red[threadIdx.x + s];
        __syncthreads();
    }
    if (threadIdx.x == 0) atomicAdd(&Tref[m], red[0]);
}

// ---------------------------------------------------------------------------
// Kernel 4: out[0] = mean((Tref - Tarr)^2) over 512 rows
// ---------------------------------------------------------------------------
__global__ __launch_bounds__(256) void k_loss(
    const float* __restrict__ Tref, const float* __restrict__ Tarr,
    float* __restrict__ out)
{
    __shared__ float red[256];
    float s = 0.0f;
    for (int i = threadIdx.x; i < MROWS; i += 256) {
        const float d = Tref[i] - Tarr[i];
        s += d * d;
    }
    red[threadIdx.x] = s;
    __syncthreads();
    for (int st = 128; st > 0; st >>= 1) {
        if (threadIdx.x < st) red[threadIdx.x] += red[threadIdx.x + st];
        __syncthreads();
    }
    if (threadIdx.x == 0) out[0] = red[0] / (float)MROWS;
}

__global__ void k_zero(float* __restrict__ p, int n) {
    const int i = blockIdx.x * blockDim.x + threadIdx.x;
    if (i < n) p[i] = 0.0f;
}

// ---------------------------------------------------------------------------
// Inputs (setup_inputs order):
// 0:d 1:x 2:ss 3:vb 4:patches 5:npp 6:sRef 7:A 8:Tarr 9:meanY 10:ds 11:lam2 12:npatches
// ---------------------------------------------------------------------------
extern "C" void kernel_launch(void* const* d_in, const int* in_sizes, int n_in,
                              void* d_out, int out_size, void* d_ws, size_t ws_size,
                              hipStream_t stream)
{
    const float* D     = (const float*)d_in[0];
    const float* X     = (const float*)d_in[1];
    const float* vb    = (const float*)d_in[3];
    const float* npp   = (const float*)d_in[5];
    const float* sRef  = (const float*)d_in[6];
    const float* A     = (const float*)d_in[7];
    const float* Tarr  = (const float*)d_in[8];
    const float* meanY = (const float*)d_in[9];
    const float* ds    = (const float*)d_in[10];
    const float* lam2  = (const float*)d_in[11];
    float* out = (float*)d_out;

    float* ws       = (float*)d_ws;
    float* ss_p_sum = ws;                       // N_SIG floats
    float* v        = ws + N_SIG;               // N_SIG floats
    float* Tref     = ws + 2 * (size_t)N_SIG;   // MROWS floats

    k_zero<<<(N_SIG + 255) / 256, 256, 0, stream>>>(ss_p_sum, N_SIG);
    k_zero<<<(MROWS + 255) / 256, 256, 0, stream>>>(Tref, MROWS);

    const int gemm_blocks = (KCOLS + COLS_PER_BLOCK - 1) / COLS_PER_BLOCK; // 2048
    k_gemm_diag<<<gemm_blocks, 256, 0, stream>>>(D, X, meanY, ss_p_sum);

    k_finalize<<<(N_SIG + 255) / 256, 256, 0, stream>>>(ss_p_sum, ds, npp, vb,
                                                        sRef, lam2, v);

    k_matvec<<<dim3(8, MROWS), 256, 0, stream>>>(A, v, Tref);

    k_loss<<<1, 256, 0, stream>>>(Tref, Tarr, out);
}
